// SAFGAT_57947698758296
// MI455X (gfx1250) — matrix operand order repaired and checked by reading, not compiled, not run on hardware
//
#include <hip/hip_runtime.h>
#include <hip/hip_bf16.h>

typedef __attribute__((ext_vector_type(16))) __bf16 v16bf;
typedef __attribute__((ext_vector_type(8)))  __bf16 v8bf;
typedef __attribute__((ext_vector_type(8)))  float  v8f;

#define LRELU_ALPHA 0.2f

// ---------- helpers ----------
__device__ __forceinline__ unsigned short f2bf(float f) {
  unsigned u = __float_as_uint(f);
  unsigned r = u + 0x7FFFu + ((u >> 16) & 1u);   // round-to-nearest-even
  return (unsigned short)(r >> 16);
}
// monotone float<->uint mapping for atomicMax on signed floats
__device__ __forceinline__ unsigned orderedF(float f) {
  unsigned u = __float_as_uint(f);
  return (u & 0x80000000u) ? ~u : (u | 0x80000000u);
}
__device__ __forceinline__ float unorderF(unsigned u) {
  return __uint_as_float((u & 0x80000000u) ? (u & 0x7FFFFFFFu) : ~u);
}
__device__ __forceinline__ v16bf joinbf(v8bf lo, v8bf hi) {
  return __builtin_shufflevector(lo, hi, 0, 1, 2, 3, 4, 5, 6, 7,
                                 8, 9, 10, 11, 12, 13, 14, 15);
}

// f32 -> bf16 bulk conversion, 4 elements/thread (count % 4 == 0)
__global__ __launch_bounds__(256)
void cvt_bf16_kernel(const float* __restrict__ src, __bf16* __restrict__ dst,
                     long long count) {
  long long i = ((long long)blockIdx.x * 256 + threadIdx.x) * 4;
  if (i >= count) return;
  float4 f = *(const float4*)(src + i);
  unsigned lo = ((unsigned)f2bf(f.y) << 16) | f2bf(f.x);
  unsigned hi = ((unsigned)f2bf(f.w) << 16) | f2bf(f.z);
  *(uint2*)(dst + i) = make_uint2(lo, hi);
}

// ELU then write bf16 (output only feeds the next GEMM), 4 elems/thread
__global__ __launch_bounds__(256)
void elu_bf16_kernel(const float* __restrict__ src, __bf16* __restrict__ dst,
                     long long count) {
  long long i = ((long long)blockIdx.x * 256 + threadIdx.x) * 4;
  if (i >= count) return;
  float4 f = *(const float4*)(src + i);
  f.x = f.x > 0.f ? f.x : (__expf(f.x) - 1.f);
  f.y = f.y > 0.f ? f.y : (__expf(f.y) - 1.f);
  f.z = f.z > 0.f ? f.z : (__expf(f.z) - 1.f);
  f.w = f.w > 0.f ? f.w : (__expf(f.w) - 1.f);
  unsigned lo = ((unsigned)f2bf(f.y) << 16) | f2bf(f.x);
  unsigned hi = ((unsigned)f2bf(f.w) << 16) | f2bf(f.z);
  *(uint2*)(dst + i) = make_uint2(lo, hi);
}

// ---------- WMMA GEMM (bf16 in, f32 out), whole-K LDS A block ----------
// Y[M,Ncols] = X[M,K] @ W[Ncols,K]^T + bias. One 16-row block per workgroup;
// the full [16,K] bf16 A panel (16KB @ K=512) is staged once with a single
// barrier, then the fully-unrolled K loop runs barrier-free from LDS.
// 8 waves each own 64 output cols (4 WMMA 16x16 subtiles).
template <int K>
__global__ __launch_bounds__(256)
void wmma_gemm_lds(const __bf16* __restrict__ X, const __bf16* __restrict__ W,
                   const float* __restrict__ bias, float* __restrict__ Y,
                   int Ncols) {
  __shared__ __align__(16) __bf16 sA[16 * K];
  const int lane = threadIdx.x & 31;
  const int wave = threadIdx.x >> 5;
  const int rowBase  = blockIdx.x * 16;
  const int colBase0 = wave * 64;

  const int mn  = lane & 15;          // A row / B col within tile
  const int kh  = (lane >> 4) * 16;   // A-frag K offset per lane-half
  const int kh2 = (lane >> 4) * 16;   // B-frag K offset per lane-half

  // Stage contiguous [16,K] A panel -> LDS (coalesced uint4 copy, 1 barrier)
  {
    const uint4* src = (const uint4*)(X + (size_t)rowBase * K);
    uint4* dst = (uint4*)sA;
#pragma unroll
    for (int i = 0; i < (16 * K) / (8 * 256); ++i)
      dst[threadIdx.x + i * 256] = src[threadIdx.x + i * 256];
  }
  __syncthreads();

  v8f acc[4] = {};
  const __bf16* arow = sA + mn * K + kh;                       // LDS base
  const __bf16* wb0 = W + (size_t)(colBase0 + 0  + mn) * K + kh2;
  const __bf16* wb1 = W + (size_t)(colBase0 + 16 + mn) * K + kh2;
  const __bf16* wb2 = W + (size_t)(colBase0 + 32 + mn) * K + kh2;
  const __bf16* wb3 = W + (size_t)(colBase0 + 48 + mn) * K + kh2;

#pragma unroll
  for (int k0 = 0; k0 < K; k0 += 32) {
    v16bf a = joinbf(*(const v8bf*)(arow + k0), *(const v8bf*)(arow + k0 + 8));
    v16bf b0 = *(const v16bf*)(wb0 + k0);
    v16bf b1 = *(const v16bf*)(wb1 + k0);
    v16bf b2 = *(const v16bf*)(wb2 + k0);
    v16bf b3 = *(const v16bf*)(wb3 + k0);
    acc[0] = __builtin_amdgcn_wmma_f32_16x16x32_bf16(false, a, false, b0, (short)0, acc[0], false, false);
    acc[1] = __builtin_amdgcn_wmma_f32_16x16x32_bf16(false, a, false, b1, (short)0, acc[1], false, false);
    acc[2] = __builtin_amdgcn_wmma_f32_16x16x32_bf16(false, a, false, b2, (short)0, acc[2], false, false);
    acc[3] = __builtin_amdgcn_wmma_f32_16x16x32_bf16(false, a, false, b3, (short)0, acc[3], false, false);
  }

  const int rowOff = (lane >> 4) * 8;
  const int colOff = lane & 15;
#pragma unroll
  for (int t = 0; t < 4; ++t) {
    const int col = colBase0 + t * 16 + colOff;
    const float bv = bias[col];
#pragma unroll
    for (int r = 0; r < 8; ++r)
      Y[(size_t)(rowBase + r + rowOff) * Ncols + col] = acc[t][r] + bv;
  }
}

// ---------- WMMA GEMM, per-wave direct-global (small Ncols: 64) ----------
template <int K>
__global__ __launch_bounds__(256)
void wmma_gemm_wave(const __bf16* __restrict__ X, const __bf16* __restrict__ W,
                    const float* __restrict__ bias, float* __restrict__ Y,
                    int Ncols, int rowTiles) {
  const int lane = threadIdx.x & 31;
  const int wave = threadIdx.x >> 5;
  const int tile = blockIdx.x * 8 + wave;
  if (tile >= rowTiles) return;                  // wave-uniform exit
  const int rowBase = tile * 16;

  const int mn  = lane & 15;
  const int kh  = (lane >> 4) * 16;
  const int kh2 = (lane >> 4) * 16;

  v8f acc[4] = {};
  const __bf16* xrow = X + (size_t)(rowBase + mn) * K + kh;
  const __bf16* wb0 = W + (size_t)(0  + mn) * K + kh2;
  const __bf16* wb1 = W + (size_t)(16 + mn) * K + kh2;
  const __bf16* wb2 = W + (size_t)(32 + mn) * K + kh2;
  const __bf16* wb3 = W + (size_t)(48 + mn) * K + kh2;

#pragma unroll
  for (int k0 = 0; k0 < K; k0 += 32) {
    v16bf a = joinbf(*(const v8bf*)(xrow + k0), *(const v8bf*)(xrow + k0 + 8));
    v16bf b0 = *(const v16bf*)(wb0 + k0);
    v16bf b1 = *(const v16bf*)(wb1 + k0);
    v16bf b2 = *(const v16bf*)(wb2 + k0);
    v16bf b3 = *(const v16bf*)(wb3 + k0);
    acc[0] = __builtin_amdgcn_wmma_f32_16x16x32_bf16(false, a, false, b0, (short)0, acc[0], false, false);
    acc[1] = __builtin_amdgcn_wmma_f32_16x16x32_bf16(false, a, false, b1, (short)0, acc[1], false, false);
    acc[2] = __builtin_amdgcn_wmma_f32_16x16x32_bf16(false, a, false, b2, (short)0, acc[2], false, false);
    acc[3] = __builtin_amdgcn_wmma_f32_16x16x32_bf16(false, a, false, b3, (short)0, acc[3], false, false);
  }

  const int rowOff = (lane >> 4) * 8;
  const int colOff = lane & 15;
#pragma unroll
  for (int t = 0; t < 4; ++t) {
    const int col = t * 16 + colOff;
    const float bv = bias[col];
#pragma unroll
    for (int r = 0; r < 8; ++r)
      Y[(size_t)(rowBase + r + rowOff) * Ncols + col] = acc[t][r] + bv;
  }
}

// ---------- attention scalar logits: f1/f2 [N,8] ----------
__global__ __launch_bounds__(256)
void f1f2_kernel(const float* __restrict__ hbuf, const float* __restrict__ a_src,
                 const float* __restrict__ a_dest, float* __restrict__ f1,
                 float* __restrict__ f2, int N) {
  long long idx = (long long)blockIdx.x * 256 + threadIdx.x;
  if (idx >= (long long)N * 8) return;
  int n = (int)(idx >> 3), h = (int)(idx & 7);
  const float* hp = hbuf + (size_t)n * 512 + h * 64;
  const float* as = a_src + h * 64;
  const float* ad = a_dest + h * 64;
  float s1 = 0.f, s2 = 0.f;
#pragma unroll 8
  for (int d = 0; d < 64; ++d) {
    float hv = hp[d];
    s1 += hv * as[d];
    s2 += hv * ad[d];
  }
  f1[idx] = s1;
  f2[idx] = s2;
}

__global__ __launch_bounds__(256)
void zero_kernel(float* __restrict__ p, long long count) {
  long long idx = (long long)blockIdx.x * 256 + threadIdx.x;
  if (idx < count) p[idx] = 0.f;
}

// ---------- edge softmax passes ----------
__global__ __launch_bounds__(256)
void edge_pass1(const float* __restrict__ f1, const float* __restrict__ f2,
                const int* __restrict__ row, const int* __restrict__ col,
                float* __restrict__ ebuf, unsigned* __restrict__ mbuf, int E) {
  long long idx = (long long)blockIdx.x * 256 + threadIdx.x;
  if (idx >= (long long)E * 8) return;
  int e = (int)(idx >> 3), h = (int)(idx & 7);
  int r = row[e], c = col[e];
  float v = f1[r * 8 + h] + f2[c * 8 + h];
  v = v > 0.f ? v : LRELU_ALPHA * v;          // LeakyReLU
  ebuf[idx] = v;
  atomicMax(&mbuf[r * 8 + h], orderedF(v));   // segment max over row
}

__global__ __launch_bounds__(256)
void edge_pass2(const int* __restrict__ row, float* __restrict__ ebuf,
                const unsigned* __restrict__ mbuf, float* __restrict__ denom, int E) {
  long long idx = (long long)blockIdx.x * 256 + threadIdx.x;
  if (idx >= (long long)E * 8) return;
  int e = (int)(idx >> 3), h = (int)(idx & 7);
  int r = row[e];
  float ex = __expf(ebuf[idx] - unorderF(mbuf[r * 8 + h]));
  ebuf[idx] = ex;
  atomicAdd(&denom[r * 8 + h], ex);           // segment sum over row
}

__global__ __launch_bounds__(256)
void edge_pass3(const int* __restrict__ row, float* __restrict__ ebuf,
                const float* __restrict__ denom, int E) {
  long long idx = (long long)blockIdx.x * 256 + threadIdx.x;
  if (idx >= (long long)E * 8) return;
  int e = (int)(idx >> 3), h = (int)(idx & 7);
  ebuf[idx] = ebuf[idx] / denom[row[e] * 8 + h];
}

// ---------- SPMM: out[row[e], :] += attn[e,h] * xw[col[e], :] ----------
__global__ __launch_bounds__(256)
void spmm_kernel(const float* __restrict__ attn, const float* __restrict__ xw,
                 const int* __restrict__ row, const int* __restrict__ col,
                 float* __restrict__ out, int E) {
  long long idx = (long long)blockIdx.x * 256 + threadIdx.x; // E*128 threads
  if (idx >= (long long)E * 128) return;
  int e  = (int)(idx >> 7);
  int c4 = (int)(idx & 127) << 2;             // 4 consecutive cols
  int h  = c4 >> 6;
  int r = row[e], c = col[e];
  float a = attn[(size_t)e * 8 + h];
  float4 x4 = *(const float4*)(xw + (size_t)c * 512 + c4);
  float* o = out + (size_t)r * 512 + c4;
  atomicAdd(o + 0, a * x4.x);
  atomicAdd(o + 1, a * x4.y);
  atomicAdd(o + 2, a * x4.z);
  atomicAdd(o + 3, a * x4.w);
}

// ---------- driver ----------
extern "C" void kernel_launch(void* const* d_in, const int* in_sizes, int n_in,
                              void* d_out, int out_size, void* d_ws, size_t ws_size,
                              hipStream_t stream) {
  const float* feat   = (const float*)d_in[0];
  const int*   eidx   = (const int*)  d_in[1];
  const float* fW_w   = (const float*)d_in[2];
  const float* fW_b   = (const float*)d_in[3];
  const float* a_src  = (const float*)d_in[4];
  const float* a_dest = (const float*)d_in[5];
  const float* W0     = (const float*)d_in[6];
  const float* b0     = (const float*)d_in[7];
  const float* W1     = (const float*)d_in[8];
  const float* b1     = (const float*)d_in[9];
  const float* lin_w  = (const float*)d_in[10];
  const float* lin_b  = (const float*)d_in[11];

  const int NF = 512;
  const int N = in_sizes[0] / NF;    // 100000 (multiple of 16)
  const int E = in_sizes[1] / 2;     // 1600000
  const int* rowp = eidx;
  const int* colp = eidx + E;

  float* ws = (float*)d_ws;
  float*    bufA  = ws;                               // [N,512] f32
  float*    bufB  = bufA + (size_t)N * 512;           // [N,512] f32
  float*    f1    = bufB + (size_t)N * 512;           // [N,8]
  float*    f2    = f1 + (size_t)N * 8;               // [N,8]
  unsigned* mbuf  = (unsigned*)(f2 + (size_t)N * 8);  // [N,8]
  float*    denom = (float*)(mbuf + (size_t)N * 8);   // [N,8]
  float*    ebuf  = denom + (size_t)N * 8;            // [E,8] attn scratch
  __bf16*   abf   = (__bf16*)(ebuf + (size_t)E * 8);  // [N,512] bf16 activations
  __bf16*   wbf0  = abf + (size_t)N * 512;            // [512,512] bf16
  __bf16*   wbf1  = wbf0 + 512 * 512;
  __bf16*   wbf2  = wbf1 + 512 * 512;
  __bf16*   wbfL  = wbf2 + 512 * 512;                 // [64,512] bf16

  dim3 blk(256);
  const int rowTiles = N / 16;
  auto cdiv = [](long long a, long long b) { return (unsigned)((a + b - 1) / b); };

  // ---- one-time precision conversion (off the GEMM hot loop) ----
  cvt_bf16_kernel<<<dim3(cdiv((long long)N * 512, 1024)), blk, 0, stream>>>(
      feat, abf, (long long)N * 512);
  cvt_bf16_kernel<<<dim3(cdiv(512 * 512, 1024)), blk, 0, stream>>>(fW_w, wbf0, 512 * 512);
  cvt_bf16_kernel<<<dim3(cdiv(512 * 512, 1024)), blk, 0, stream>>>(W0, wbf1, 512 * 512);
  cvt_bf16_kernel<<<dim3(cdiv(512 * 512, 1024)), blk, 0, stream>>>(W1, wbf2, 512 * 512);
  cvt_bf16_kernel<<<dim3(cdiv(64 * 512, 1024)), blk, 0, stream>>>(lin_w, wbfL, 64 * 512);

  // h = feat @ fW_w^T + fW_b    [N,512]
  wmma_gemm_lds<512><<<dim3(rowTiles), blk, 0, stream>>>(abf, wbf0, fW_b, bufA, 512);
  // f1/f2 attention logits
  f1f2_kernel<<<dim3(cdiv((long long)N * 8, 256)), blk, 0, stream>>>(bufA, a_src, a_dest, f1, f2, N);
  // zero max + denom (contiguous, bit-pattern 0 valid for both)
  zero_kernel<<<dim3(cdiv((long long)N * 16, 256)), blk, 0, stream>>>((float*)mbuf, (long long)N * 16);
  // edge softmax
  unsigned egrid = cdiv((long long)E * 8, 256);
  edge_pass1<<<dim3(egrid), blk, 0, stream>>>(f1, f2, rowp, colp, ebuf, mbuf, E);
  edge_pass2<<<dim3(egrid), blk, 0, stream>>>(rowp, ebuf, mbuf, denom, E);
  edge_pass3<<<dim3(egrid), blk, 0, stream>>>(rowp, ebuf, denom, E);

  // layer 0: xw0 = feat @ W0^T + b0 -> bufA ; agg -> bufB ; ELU -> abf (bf16)
  wmma_gemm_lds<512><<<dim3(rowTiles), blk, 0, stream>>>(abf, wbf1, b0, bufA, 512);
  zero_kernel<<<dim3(cdiv((long long)N * 512, 256)), blk, 0, stream>>>(bufB, (long long)N * 512);
  spmm_kernel<<<dim3(cdiv((long long)E * 128, 256)), blk, 0, stream>>>(ebuf, bufA, rowp, colp, bufB, E);
  elu_bf16_kernel<<<dim3(cdiv((long long)N * 512, 1024)), blk, 0, stream>>>(
      bufB, abf, (long long)N * 512);

  // layer 1: xw1 = x1 @ W1^T + b1 -> bufA ; agg -> bufB ; ELU -> abf (bf16)
  wmma_gemm_lds<512><<<dim3(rowTiles), blk, 0, stream>>>(abf, wbf2, b1, bufA, 512);
  zero_kernel<<<dim3(cdiv((long long)N * 512, 256)), blk, 0, stream>>>(bufB, (long long)N * 512);
  spmm_kernel<<<dim3(cdiv((long long)E * 128, 256)), blk, 0, stream>>>(ebuf, bufA, rowp, colp, bufB, E);
  elu_bf16_kernel<<<dim3(cdiv((long long)N * 512, 1024)), blk, 0, stream>>>(
      bufB, abf, (long long)N * 512);

  // output projection: out = x2 @ lin_w^T + lin_b   [N,64]
  wmma_gemm_wave<512><<<dim3(cdiv(rowTiles, 8), 1), blk, 0, stream>>>(
      abf, wbfL, lin_b, (float*)d_out, 64, rowTiles);
}
